// DenoisingConvNN_2D_Spatial_K_N_25039659335748
// MI455X (gfx1250) — matrix operand order repaired and checked by reading, not compiled, not run on hardware
//
#include <hip/hip_runtime.h>
#include <hip/hip_bf16.h>

#define TK     9        // neighbors (K in reference)
#define NANCH  64       // anchor count (N*N)
#define HH     128
#define WW     128
#define TPIX   (HH*WW)
#define BATCH  8

typedef __attribute__((ext_vector_type(16))) _Float16 v16h;
typedef __attribute__((ext_vector_type(8)))  float    v8f;
typedef __attribute__((ext_vector_type(4))) unsigned int u32x4;
typedef __attribute__((ext_vector_type(8))) int          i32x8;
typedef __attribute__((ext_vector_type(4))) int          i32x4;

// ---------------- pixel_unshuffle: x(B,3,256,256) -> u0[b][t][12] (pixel-major) ----
__global__ void unshuffle_kernel(const float* __restrict__ x, float* __restrict__ u0) {
  int i = blockIdx.x * blockDim.x + threadIdx.x;
  const int total = BATCH * TPIX * 12;
  if (i >= total) return;
  int cc = i % 12;
  int t  = (i / 12) % TPIX;
  int b  = i / (12 * TPIX);
  int c0 = cc >> 2, s1 = (cc >> 1) & 1, s2 = cc & 1;
  int h = t >> 7, w = t & 127;
  u0[i] = x[(((size_t)b * 3 + c0) * 256 + (2 * h + s1)) * 256 + (2 * w + s2)];
}

// ---------------- anchors: gather xs[b][c][64], squared norms, and f16 copy -------
__global__ void anchors_kernel(const float* __restrict__ F, int C, int Cp,
                               float* __restrict__ xs, float* __restrict__ s2n,
                               _Float16* __restrict__ XSh) {
  int bm = blockIdx.x * blockDim.x + threadIdx.x;
  if (bm >= BATCH * NANCH) return;
  int b = bm >> 6, m = bm & 63;
  int mi = m >> 3, mj = m & 7;
  int h = (int)roundf(mi * 127.0f / 7.0f);   // matches np.round(np.linspace(0,127,8))
  int w = (int)roundf(mj * 127.0f / 7.0f);
  int t = h * WW + w;
  const float* xv = F + ((size_t)b * TPIX + t) * C;
  float s = 0.f;
  for (int c = 0; c < C; ++c) {
    float v = xv[c];
    s += v * v;
    xs [((size_t)b * C  + c) * NANCH + m] = v;
    XSh[((size_t)b * Cp + c) * NANCH + m] = (_Float16)v;
  }
  for (int c = C; c < Cp; ++c) XSh[((size_t)b * Cp + c) * NANCH + m] = (_Float16)0.f;
  s2n[b * NANCH + m] = s;
}

// ---------------- weights f32 (O,C,TK) -> f16 padded Wh[k][o][c] ------------------
__global__ void convw_kernel(const float* __restrict__ w, int O, int C, int Op, int Cp,
                             _Float16* __restrict__ Wh) {
  int i = blockIdx.x * blockDim.x + threadIdx.x;
  int total = TK * Op * Cp;
  if (i >= total) return;
  int c = i % Cp;
  int o = (i / Cp) % Op;
  int k = i / (Cp * Op);
  float v = (o < O && c < C) ? w[((size_t)o * C + c) * TK + k] : 0.f;
  Wh[i] = (_Float16)v;
}

// ---------------- WMMA: P_t[b][k][anchor m][o] = sum_c Wh[k][o][c] * XSh[b][c][m] --
// One wave per 16x16 D tile; CDNA5 v_wmma_f32_16x16x32_f16, f32 accumulate.
template <int CP>
__global__ void __launch_bounds__(32)
wmma_P_kernel(const _Float16* __restrict__ Wh, const _Float16* __restrict__ XSh,
              float* __restrict__ P, int Op) {
  const int lane = threadIdx.x & 31;
  const int hi   = lane >> 4;       // 0 or 1
  const int l16  = lane & 15;
  const int nt = blockIdx.x & 3;    // anchor tile (64/16 = 4)
  const int mt = blockIdx.x >> 2;   // output-channel tile
  const int kk = blockIdx.y;        // 0..TK-1
  const int b  = blockIdx.z;        // batch
  const int obase = mt * 16;
  const int mbase = nt * 16;
  const int kb = hi ? 8 : 0;

  v8f acc = {};
  #pragma unroll
  for (int cb = 0; cb < CP; cb += 32) {
    const _Float16* Arow = Wh + ((size_t)kk * Op + obase + l16) * CP + cb;
    v16h a;
    #pragma unroll
    for (int e = 0; e < 8; ++e) { a[e] = Arow[kb + e]; a[8 + e] = Arow[kb + 16 + e]; }
    const _Float16* Bcol = XSh + ((size_t)b * CP + cb + (hi ? 8 : 0)) * NANCH + mbase + l16;
    v16h bf;
    #pragma unroll
    for (int e = 0; e < 16; ++e) bf[e] = Bcol[(size_t)(e < 8 ? e : e + 8) * NANCH];
    acc = __builtin_amdgcn_wmma_f32_16x16x32_f16(
        /*neg_a=*/false, a, /*neg_b=*/false, bf,
        /*c_mod=*/(short)0, acc, /*reuse_a=*/false, /*reuse_b=*/false);
  }
  #pragma unroll
  for (int r = 0; r < 8; ++r) {
    int o = obase + r + hi * 8;
    int m = mbase + l16;
    P[(((size_t)b * TK + kk) * NANCH + m) * Op + o] = acc[r];
  }
}

// ---------------- per-pixel: distances -> wave top-9 -> coalesced P gather --------
__global__ void __launch_bounds__(256)
pixel_kernel(const float* __restrict__ F, int C,
             const float* __restrict__ xs, const float* __restrict__ s2n,
             const float* __restrict__ P, int Op, int O,
             const float* __restrict__ bias, int relu,
             float* __restrict__ Y) {
  __shared__ float sh_xs[128 * NANCH];   // max C = 128 -> 32 KB
  __shared__ float sh_s2[NANCH];
  const int b = blockIdx.y;
  const int nElem = C * NANCH;           // <= 8192 f32 elements
  const float* xsb = xs + (size_t)b * nElem;

#if defined(__gfx1250__) && __has_builtin(__builtin_amdgcn_tensor_load_to_lds)
  // ---- Tensor Data Mover: DMA xs[b] (C*64 f32) into sh_xs. Wave 0 issues once. ----
  if (threadIdx.x < 32) {
    unsigned long long ga = (unsigned long long)(const void*)xsb;
    unsigned ldsoff = (unsigned)(unsigned long long)(const void*)sh_xs; // LDS offset = low 32 bits
    unsigned n = (unsigned)nElem;
    u32x4 g0;
    g0[0] = 1u;                                      // count=1, user descriptor
    g0[1] = ldsoff;                                  // lds_addr
    g0[2] = (unsigned)ga;                            // global_addr[31:0]
    g0[3] = (unsigned)((ga >> 32) & 0x1FFFFFFull) | (2u << 30);  // addr[56:32] | type=2
    i32x8 g1;
    g1[0] = (int)(2u << 16);                         // data_size=2 (4 bytes), mask=0
    g1[1] = (int)((n & 0xFFFFu) << 16);              // tensor_dim0[15:0] at bit 48
    g1[2] = (int)(((n >> 16) & 0xFFFFu) | (1u << 16)); // tensor_dim0[31:16] | tensor_dim1=1 (lo)
    g1[3] = (int)((n & 0xFFFFu) << 16);              // tensor_dim1 hi=0 | tile_dim0=n
    g1[4] = 0;                                       // tile_dim1=0 (unused), tile_dim2=0
    g1[5] = (int)n;                                  // tensor_dim0_stride lo (benign for 1-D)
    g1[6] = 0;
    g1[7] = 0;
    i32x4 z4 = {0, 0, 0, 0};
    i32x8 z8 = {0, 0, 0, 0, 0, 0, 0, 0};
    __builtin_amdgcn_tensor_load_to_lds(g0, g1, z4, z4, z8, 0);  // 6-arg toolchain form
#if __has_builtin(__builtin_amdgcn_s_wait_tensorcnt)
    __builtin_amdgcn_s_wait_tensorcnt(0);
#else
    asm volatile("s_wait_tensorcnt 0x0" ::: "memory");
#endif
  }
  __syncthreads();
#endif
  // Idempotent staging pass (hedge: rewrites identical data; repairs any DMA slip).
  for (int i = threadIdx.x; i < nElem; i += blockDim.x)
    sh_xs[i] = xsb[i];

#if defined(__gfx1250__) && __has_builtin(__builtin_amdgcn_global_load_async_to_lds_b32)
  if (threadIdx.x < NANCH) {
    __builtin_amdgcn_global_load_async_to_lds_b32(
        (__attribute__((address_space(1))) int*)(unsigned long long)(const void*)
            (s2n + b * NANCH + threadIdx.x),
        (__attribute__((address_space(3))) int*)(unsigned long long)(const void*)
            (&sh_s2[threadIdx.x]),
        0, 0);
  }
#if __has_builtin(__builtin_amdgcn_s_wait_asynccnt)
  __builtin_amdgcn_s_wait_asynccnt(0);
#else
  asm volatile("s_wait_asynccnt 0x0" ::: "memory");
#endif
#else
  if (threadIdx.x < NANCH) sh_s2[threadIdx.x] = s2n[b * NANCH + threadIdx.x];
#endif
  __syncthreads();

  const int lane   = threadIdx.x & 31;
  const int nwaves = blockDim.x >> 5;
  const int wave   = blockIdx.x * nwaves + (threadIdx.x >> 5);
  const int stride = gridDim.x * nwaves;
  const float* Pb = P + (size_t)b * TK * NANCH * Op;

  for (int t = wave; t < TPIX; t += stride) {
    // distances to 64 anchors; ||x_t||^2 dropped (rank-invariant)
    const float* xv = F + ((size_t)b * TPIX + t) * C;
    float a0 = 0.f, a1 = 0.f;
    for (int c = 0; c < C; ++c) {
      float x = xv[c];                         // uniform across wave -> broadcast
      a0 = fmaf(x, sh_xs[c * NANCH + lane], a0);
      a1 = fmaf(x, sh_xs[c * NANCH + lane + 32], a1);
    }
    float d0 = sh_s2[lane]      - 2.f * a0;
    float d1 = sh_s2[lane + 32] - 2.f * a1;

    // wave-cooperative top-9 (min distance, ties -> lower anchor index)
    unsigned used = 0;
    int nbr[TK];
    #pragma unroll
    for (int kk = 0; kk < TK; ++kk) {
      float v0 = (used & 1u) ? 3.0e38f : d0;
      float v1 = (used & 2u) ? 3.0e38f : d1;
      float bv; int bi;
      if (v0 <= v1) { bv = v0; bi = lane; } else { bv = v1; bi = lane + 32; }
      #pragma unroll
      for (int off = 16; off > 0; off >>= 1) {
        float ov = __shfl_xor(bv, off, 32);
        int   oi = __shfl_xor(bi, off, 32);
        if (ov < bv || (ov == bv && oi < bi)) { bv = ov; bi = oi; }
      }
      nbr[kk] = bi;                               // uniform across wave
      if ((bi & 31) == lane) used |= (bi < 32) ? 1u : 2u;
    }

    #pragma unroll
    for (int kk = 0; kk < TK; ++kk)               // global_prefetch_b8 of P rows
      __builtin_prefetch(Pb + ((size_t)kk * NANCH + nbr[kk]) * Op, 0, 1);

    // y[o] = bias[o] + sum_k P_t[k][nbr_k][o]  (lanes -> consecutive o, coalesced)
    for (int o = lane; o < O; o += 32) {
      float acc = bias[o];
      #pragma unroll
      for (int kk = 0; kk < TK; ++kk)
        acc += Pb[((size_t)kk * NANCH + nbr[kk]) * Op + o];
      if (relu) acc = fmaxf(acc, 0.f);
      Y[((size_t)b * TPIX + t) * O + o] = acc;
    }
  }
}

// ---------------- final pixel_shuffle: y3[b][t][12] -> out(B,3,256,256) -----------
__global__ void shuffle_kernel(const float* __restrict__ y3, float* __restrict__ out) {
  int i = blockIdx.x * blockDim.x + threadIdx.x;
  const int total = BATCH * 3 * 256 * 256;
  if (i >= total) return;
  int ww2 = i % 256;
  int hh2 = (i / 256) % 256;
  int c0  = (i / (256 * 256)) % 3;
  int b   = i / (3 * 256 * 256);
  int h = hh2 >> 1, s1 = hh2 & 1;
  int w = ww2 >> 1, s2 = ww2 & 1;
  out[i] = y3[((size_t)b * TPIX + (h * WW + w)) * 12 + (c0 * 4 + s1 * 2 + s2)];
}

extern "C" void kernel_launch(void* const* d_in, const int* in_sizes, int n_in,
                              void* d_out, int out_size, void* d_ws, size_t ws_size,
                              hipStream_t stream) {
  (void)in_sizes; (void)n_in; (void)out_size; (void)ws_size;
  const float* x  = (const float*)d_in[0];
  const float* w1 = (const float*)d_in[1];
  const float* b1 = (const float*)d_in[2];
  const float* w2 = (const float*)d_in[3];
  const float* b2 = (const float*)d_in[4];
  const float* w3 = (const float*)d_in[5];
  const float* b3 = (const float*)d_in[6];
  float* out = (float*)d_out;

  char* ws = (char*)d_ws;
  size_t off = 0;
  auto alloc = [&](size_t bytes) {
    void* p = ws + off;
    off = (off + bytes + 255) & ~(size_t)255;
    return p;
  };
  float*    u0  = (float*)   alloc((size_t)BATCH * TPIX * 12  * 4);
  float*    y1  = (float*)   alloc((size_t)BATCH * TPIX * 64  * 4);
  float*    y2  = (float*)   alloc((size_t)BATCH * TPIX * 128 * 4);
  float*    y3  = (float*)   alloc((size_t)BATCH * TPIX * 12  * 4);
  float*    xs  = (float*)   alloc((size_t)BATCH * 128 * NANCH * 4);
  float*    s2n = (float*)   alloc((size_t)BATCH * NANCH * 4);
  _Float16* XSh = (_Float16*)alloc((size_t)BATCH * 128 * NANCH * 2);
  _Float16* Wh  = (_Float16*)alloc((size_t)TK * 128 * 64 * 2);        // max Op*Cp=8192
  float*    P   = (float*)   alloc((size_t)BATCH * TK * NANCH * 128 * 4);

  { int total = BATCH * TPIX * 12;
    unshuffle_kernel<<<(total + 255) / 256, 256, 0, stream>>>(x, u0); }

  auto run_layer = [&](const float* F, int C, int Cp, int O, int Op,
                       const float* W, const float* bias, float* Y, int relu) {
    anchors_kernel<<<(BATCH * NANCH + 255) / 256, 256, 0, stream>>>(F, C, Cp, xs, s2n, XSh);
    int totw = TK * Op * Cp;
    convw_kernel<<<(totw + 255) / 256, 256, 0, stream>>>(W, O, C, Op, Cp, Wh);
    dim3 g((Op / 16) * 4, TK, BATCH);
    if (Cp == 32)       wmma_P_kernel<32> <<<g, 32, 0, stream>>>(Wh, XSh, P, Op);
    else if (Cp == 64)  wmma_P_kernel<64> <<<g, 32, 0, stream>>>(Wh, XSh, P, Op);
    else                wmma_P_kernel<128><<<g, 32, 0, stream>>>(Wh, XSh, P, Op);
    dim3 gp(256, BATCH);
    pixel_kernel<<<gp, 256, 0, stream>>>(F, C, xs, s2n, P, Op, O, bias, relu, Y);
  };

  run_layer(u0, 12,  32,  64,  64, w1, b1, y1, 1);
  run_layer(y1, 64,  64, 128, 128, w2, b2, y2, 1);
  run_layer(y2, 128, 128, 12,  16, w3, b3, y3, 0);

  { int total = BATCH * 3 * 256 * 256;
    shuffle_kernel<<<(total + 255) / 256, 256, 0, stream>>>(y3, out); }
}
